// ANet_90752658964601
// MI455X (gfx1250) — matrix operand order repaired and checked by reading, not compiled, not run on hardware
//
#include <hip/hip_runtime.h>

// ---------------------------------------------------------------------------
// ConvLSTM (T=4096, 2 layers, 40x50 state) + FC(2000x2000) for MI455X gfx1250.
//
//  * Kernel 1: one persistent workgroup (16 waves). h state lives in LDS with
//    a zero halo (42x52); c state lives in per-lane registers (static
//    position->lane ownership). Per step, gates = (4x18)@(18x2000) GEMM done
//    as 5 chained V_WMMA_F32_16X16X4_F32 per 16-position tile (K=4 packs two
//    3x3 taps x two channels). Activations via gfx1250 V_TANH_F32.
//    x_t streamed HBM->LDS with GLOBAL_LOAD_ASYNC_TO_LDS_B32 (ASYNCcnt).
//  * Kernel 2: FC via V_WMMA_F32_16X16X32_F16, 125 one-wave blocks, f32 acc.
// ---------------------------------------------------------------------------

typedef float    v2f  __attribute__((ext_vector_type(2)));
typedef float    v8f  __attribute__((ext_vector_type(8)));
typedef _Float16 v16h __attribute__((ext_vector_type(16)));

#define T_STEPS   4096
#define IN_RAW    1910
#define NPOS      2000
#define PH        42
#define PW        52
#define PSZ       (PH * PW)      // 2184 padded floats
#define NTILES    125            // 2000 / 16
#define NWAVES    16
#define NTHREADS  (NWAVES * 32)
#define MAXTPW    8              // max tiles per wave (125/16 rounded up)

// gfx1250 hardware tanh: one TRANS op instead of exp+rcp chains.
__device__ __forceinline__ float ftanh(float x) {
    return __builtin_amdgcn_tanhf(x);
}
// sigmoid(x) = 0.5*tanh(x/2) + 0.5  (1 trans + 1 fma)
__device__ __forceinline__ float fsig(float x) {
    return __builtin_fmaf(0.5f, __builtin_amdgcn_tanhf(0.5f * x), 0.5f);
}

// ---------------------------------------------------------------------------
// Kernel 1: persistent ConvLSTM, single workgroup.
// ---------------------------------------------------------------------------
__global__ __launch_bounds__(NTHREADS) void convlstm_kernel(
    const float* __restrict__ S,
    const float* __restrict__ W0, const float* __restrict__ B0,
    const float* __restrict__ W1, const float* __restrict__ B1,
    float* __restrict__ feat_out)
{
    __shared__ float xb[PSZ];          // padded x_t
    __shared__ float h0b[2][PSZ];      // padded h, layer 0, ping-pong
    __shared__ float h1b[2][PSZ];      // padded h, layer 1, ping-pong

    const int tid  = threadIdx.x;
    const int lane = tid & 31;
    const int wv   = tid >> 5;
    const bool hiHalf = (lane >= 16);

    // Zero LDS (halos stay zero forever; interiors get overwritten each step).
    for (int i = tid; i < PSZ; i += NTHREADS) {
        xb[i] = 0.f; h0b[0][i] = 0.f; h0b[1][i] = 0.f; h1b[0][i] = 0.f; h1b[1][i] = 0.f;
    }

    // --- Precompute A-matrix (weight) fragments: 16x4 f32, per K-pair q. ---
    // K layout: k0=(off 2q, ch x), k1=(off 2q, ch h), k2=(off 2q+1, ch x),
    //           k3=(off 2q+1, ch h).  Lanes 0-15 hold K=0,1; lanes 16-31 K=2,3.
    // conv_w layout (4,2,3,3): w[gate][ch][off] = W[gate*18 + ch*9 + off].
    v2f a0[5], a1[5];
    {
        const int m = lane & 15;
        #pragma unroll
        for (int q = 0; q < 5; ++q) {
            const int off = 2 * q + (hiHalf ? 1 : 0);
            float x0 = 0.f, y0 = 0.f, x1 = 0.f, y1 = 0.f;
            if (m < 4 && off < 9) {
                x0 = W0[m * 18 + off];      y0 = W0[m * 18 + 9 + off];
                x1 = W1[m * 18 + off];      y1 = W1[m * 18 + 9 + off];
            }
            a0[q].x = x0; a0[q].y = y0;
            a1[q].x = x1; a1[q].y = y1;
        }
    }

    // Accumulator init templates: D rows 0-3 (lanes 0-15) carry the gate bias.
    v8f cinit0 = {}; v8f cinit1 = {};
    if (lane < 16) {
        #pragma unroll
        for (int r = 0; r < 4; ++r) { cinit0[r] = B0[r]; cinit1[r] = B1[r]; }
    }

    // --- Tile geometry, hoisted out of the time loop (no int div inside). ---
    int ntile = 0, ibase_pre[MAXTPW], p_pre[MAXTPW];
    #pragma unroll
    for (int i = 0; i < MAXTPW; ++i) { ibase_pre[i] = 0; p_pre[i] = 0; }
    for (int tI = wv; tI < NTILES; tI += NWAVES) {
        const int p = tI * 16 + (lane & 15);
        const int y = p / 50, xx = p % 50;
        ibase_pre[ntile] = (y + 1) * PW + (xx + 1);
        p_pre[ntile]     = p;
        ++ntile;
    }
    // Per-lane LDS delta for each K-pair q (this lane supplies offsets 2q or 2q+1).
    int dq[5];
    #pragma unroll
    for (int q = 0; q < 5; ++q) {
        const int off = 2 * q + (hiHalf ? 1 : 0);
        const int o   = (off < 9) ? off : 4;            // off==9 is padding (zeroed below)
        dq[q] = (o / 3 - 1) * PW + (o % 3 - 1);
    }

    // Cell state lives in registers: lane<16 owns position p_pre[ti] forever.
    float c0r[MAXTPW], c1r[MAXTPW];
    #pragma unroll
    for (int i = 0; i < MAXTPW; ++i) { c0r[i] = 0.f; c1r[i] = 0.f; }

    // x_t load slots: each thread owns fixed raw indices (zero-pad 1910->2000).
    // Precompute raw index and LDS byte offset (addrspace(3)) per slot for the
    // async HBM->LDS path.
    int xk[4]; unsigned xlds[4]; int nx = 0;
    for (int i = 0; i < 4; ++i) {
        const int k = tid + i * NTHREADS;
        if (k < IN_RAW) {
            const int y = k / 50, xx = k % 50;
            xk[nx] = k;
            float* gp = &xb[(y + 1) * PW + (xx + 1)];
            xlds[nx] = (unsigned)(unsigned long long)
                       (__attribute__((address_space(3))) float*)gp;
            ++nx;
        }
    }

    __syncthreads();

    for (int t = 0; t < T_STEPS; ++t) {
        const int cur = t & 1, nxt = cur ^ 1;

        // Phase 0: async-stream x_t into padded LDS (no VGPR round trip);
        // prefetch next step's slice toward L2.
        for (int i = 0; i < nx; ++i) {
            const float* ga = S + (long)t * IN_RAW + xk[i];
            asm volatile("global_load_async_to_lds_b32 %0, %1, off"
                         :: "v"(xlds[i]), "v"(ga) : "memory");
        }
        if (t + 1 < T_STEPS) __builtin_prefetch(&S[(long)(t + 1) * IN_RAW + tid], 0, 1);
        asm volatile("s_wait_asynccnt 0x0" ::: "memory");
        __syncthreads();

        // Phase 1: layer 0.  gates(4x16 tile) = sum_q A0[q] @ B[q] + bias.
        #pragma unroll
        for (int ti = 0; ti < MAXTPW; ++ti) {
            if (ti < ntile) {                       // wave-uniform guard
                const int ib = ibase_pre[ti];
                v8f acc = cinit0;
                #pragma unroll
                for (int q = 0; q < 5; ++q) {
                    const int a = ib + dq[q];
                    float bx = xb[a];
                    float by = h0b[cur][a];
                    if (q == 4) { bx = hiHalf ? 0.f : bx; by = hiHalf ? 0.f : by; }
                    v2f b; b.x = bx; b.y = by;
                    acc = __builtin_amdgcn_wmma_f32_16x16x4_f32(
                              false, a0[q], false, b, (short)0, acc, false, false);
                }
                if (lane < 16) {
                    const float ig = fsig(acc[0]), fg = fsig(acc[1]);
                    const float og = fsig(acc[2]), gg = ftanh(acc[3]);
                    const float cn = fg * c0r[ti] + ig * gg;
                    c0r[ti] = cn;
                    h0b[nxt][ib] = og * ftanh(cn);
                }
            }
        }
        __syncthreads();

        // Phase 2: layer 1 (x-channel = fresh h0, h-channel = previous h1).
        #pragma unroll
        for (int ti = 0; ti < MAXTPW; ++ti) {
            if (ti < ntile) {
                const int ib = ibase_pre[ti];
                v8f acc = cinit1;
                #pragma unroll
                for (int q = 0; q < 5; ++q) {
                    const int a = ib + dq[q];
                    float bx = h0b[nxt][a];
                    float by = h1b[cur][a];
                    if (q == 4) { bx = hiHalf ? 0.f : bx; by = hiHalf ? 0.f : by; }
                    v2f b; b.x = bx; b.y = by;
                    acc = __builtin_amdgcn_wmma_f32_16x16x4_f32(
                              false, a1[q], false, b, (short)0, acc, false, false);
                }
                if (lane < 16) {
                    const float ig = fsig(acc[0]), fg = fsig(acc[1]);
                    const float og = fsig(acc[2]), gg = ftanh(acc[3]);
                    const float cn = fg * c1r[ti] + ig * gg;
                    c1r[ti] = cn;
                    h1b[nxt][ib] = og * ftanh(cn);
                }
            }
        }
        __syncthreads();
    }

    // t=4095 wrote h1b[0].  feat = leaky_relu(h1_final).
    for (int k = tid; k < NPOS; k += NTHREADS) {
        const int y = k / 50, xx = k % 50;
        const float v = h1b[0][(y + 1) * PW + (xx + 1)];
        feat_out[k] = v > 0.f ? v : 0.01f * v;
    }
}

// ---------------------------------------------------------------------------
// Kernel 2: out = sigmoid(feat @ fc_w^T + fc_b) via V_WMMA_F32_16X16X32_F16.
// One wave per 16 output rows; A row 0 = feat chunk, B = 16 weight rows.
// ---------------------------------------------------------------------------
__global__ __launch_bounds__(32) void fc_kernel(
    const float* __restrict__ feat, const float* __restrict__ FW,
    const float* __restrict__ FB, float* __restrict__ out)
{
    const int lane = threadIdx.x & 31;
    const int n0   = blockIdx.x * 16;
    const int row  = n0 + (lane & 15);
    const long rbase = (long)row * 2000;
    const int kbL  = (lane >= 16) ? 8 : 0;

    v8f acc = {};
    for (int kc = 0; kc < 63; ++kc) {            // ceil(2000/32)
        const int k0 = kc * 32;

        // A (16x32 f16): row 0 = feat, rows 1-15 zero.
        // Lane 0 carries K 0-7 / 16-23; lane 16 carries K 8-15 / 24-31.
        v16h af = {};
        if (lane == 0) {
            #pragma unroll
            for (int j = 0; j < 8; ++j) {
                const int ka = k0 + j, kb = k0 + 16 + j;
                af[j]     = (_Float16)(ka < 2000 ? feat[ka] : 0.f);
                af[8 + j] = (_Float16)(kb < 2000 ? feat[kb] : 0.f);
            }
        } else if (lane == 16) {
            #pragma unroll
            for (int j = 0; j < 8; ++j) {
                const int ka = k0 + 8 + j, kb = k0 + 24 + j;
                af[j]     = (_Float16)(ka < 2000 ? feat[ka] : 0.f);
                af[8 + j] = (_Float16)(kb < 2000 ? feat[kb] : 0.f);
            }
        }

        // B (32x16 f16): lane l<16 -> col l, K 0-7 / 16-23; lanes 16-31 -> K 8-15 / 24-31.
        v16h bf;
        #pragma unroll
        for (int j = 0; j < 16; ++j) {
            const int k = k0 + kbL + j + (j >= 8 ? 8 : 0);
            bf[j] = (_Float16)(k < 2000 ? FW[rbase + k] : 0.f);
        }

        acc = __builtin_amdgcn_wmma_f32_16x16x32_f16(
                  false, af, false, bf, (short)0, acc, false, false);
    }

    // D row M=0 lives in acc[0] on lanes 0-15 (N = lane).
    if (lane < 16) {
        const float z = acc[0] + FB[n0 + lane];
        out[n0 + lane] = fsig(z);
    }
}

// ---------------------------------------------------------------------------
extern "C" void kernel_launch(void* const* d_in, const int* in_sizes, int n_in,
                              void* d_out, int out_size, void* d_ws, size_t ws_size,
                              hipStream_t stream)
{
    const float* S  = (const float*)d_in[0];   // (1,4096,1,1,1910)
    const float* W0 = (const float*)d_in[1];   // (4,2,3,3)
    const float* B0 = (const float*)d_in[2];   // (4,)
    const float* W1 = (const float*)d_in[3];
    const float* B1 = (const float*)d_in[4];
    const float* FW = (const float*)d_in[5];   // (2000,2000)
    const float* FB = (const float*)d_in[6];   // (2000,)
    float* out  = (float*)d_out;               // (1,2000)
    float* feat = (float*)d_ws;                // 2000 floats of scratch

    convlstm_kernel<<<1, NTHREADS, 0, stream>>>(S, W0, B0, W1, B1, feat);
    fc_kernel<<<125, 32, 0, stream>>>(feat, FW, FB, out);
    (void)in_sizes; (void)n_in; (void)out_size; (void)ws_size;
}
